// SparseGraphAttentionLayer_89111981457424
// MI455X (gfx1250) — matrix operand order repaired and checked by reading, not compiled, not run on hardware
//
#include <hip/hip_runtime.h>

#define D_IN  256
#define D_OUT 128
#define ALPHA 0.2f

typedef __attribute__((ext_vector_type(16))) _Float16 v16h;
typedef __attribute__((ext_vector_type(8)))  _Float16 v8h;
typedef __attribute__((ext_vector_type(4)))  _Float16 v4h;
typedef __attribute__((ext_vector_type(8)))  float    v8f;

// ---------------------------------------------------------------------------
// Kernel 0: zero-init the atomic accumulation buffers (d_out, rowsum).
// ---------------------------------------------------------------------------
__global__ void k_zero(float* __restrict__ out, float* __restrict__ rowsum,
                       int n_out, int n_row) {
    int stride = gridDim.x * blockDim.x;
    for (int i = blockIdx.x * blockDim.x + threadIdx.x; i < n_out; i += stride)
        out[i] = 0.0f;
    for (int i = blockIdx.x * blockDim.x + threadIdx.x; i < n_row; i += stride)
        rowsum[i] = 0.0f;
}

// ---------------------------------------------------------------------------
// Kernel 1: h = x @ W via v_wmma_f32_16x16x32_f16.
// Block = 256 threads = 8 waves. Block tile = 128 rows x 128 cols.
// Each wave owns one 16-row strip (8 accumulators of 16x16).
// LDS layout chosen so every lane's fragment is a contiguous, 16B-aligned
// run -> ds_load_b128 instead of scalar ds_load_u16 gathers:
//   lx : row-major [row][k]      (A: two contiguous 8-half K-runs per lane)
//   lwT: TRANSPOSED [n][k]       (B: one contiguous 16-half K-run per lane)
// ---------------------------------------------------------------------------
__global__ __launch_bounds__(256) void k_gemm(const float* __restrict__ x,
                                              const float* __restrict__ W,
                                              float* __restrict__ h, int N) {
    __shared__ _Float16 lx [128 * 32];  // 128 rows x 32 k        (8 KB)
    __shared__ _Float16 lwT[128 * 32];  // 128 cols x 32 k (transposed, 8 KB)

    const int tid     = threadIdx.x;
    const int wave    = tid >> 5;
    const int lane    = tid & 31;
    const int n16     = lane & 15;      // lane within half-wave
    const int hi      = lane >> 4;      // 0 = lanes 0-15, 1 = lanes 16-31
    const int m0      = wave * 16;      // wave's row strip inside the block
    const int rowBase = blockIdx.x * 128;

    v8f acc[8] = {};                    // 8 n-tiles of 16x16 f32

    for (int kb = 0; kb < D_IN; kb += 32) {
        // ---- stage x[128x32] as f16, float4 global loads, v4h LDS stores ----
        for (int i = 0; i < 4; ++i) {
            int j   = i * 256 + tid;            // float4 index, 0..1023
            int r   = j >> 3;                   // row in tile (8 float4/row)
            int c4  = (j & 7) * 4;              // k offset
            int row = rowBase + r;
            float4 xv = make_float4(0.f, 0.f, 0.f, 0.f);
            if (row < N)
                xv = *(const float4*)&x[(size_t)row * D_IN + kb + c4];
            v4h p;
            p[0] = (_Float16)xv.x; p[1] = (_Float16)xv.y;
            p[2] = (_Float16)xv.z; p[3] = (_Float16)xv.w;
            *(v4h*)&lx[r * 32 + c4] = p;        // 8B aligned
        }
        // ---- stage W[32x128] TRANSPOSED as f16: lwT[n][k] ----
        // thread owns (n = tid&127, k-halfrun = (tid>>7)*16): coalesced global
        // reads per k, one contiguous 32B LDS store (2x ds_store_b128).
        {
            int n  = tid & 127;
            int kg = (tid >> 7) * 16;
            v16h wp;
            for (int kk = 0; kk < 16; ++kk)
                wp[kk] = (_Float16)W[(size_t)(kb + kg + kk) * D_OUT + n];
            *(v16h*)&lwT[n * 32 + kg] = wp;     // 32B aligned
        }
        __syncthreads();

        // prefetch next k-slab of x into cache (global_prefetch_b8)
        if (kb + 32 < D_IN) {
            int prow = rowBase + (tid >> 1);
            if (prow < N)
                __builtin_prefetch(&x[(size_t)prow * D_IN + kb + 32 + (tid & 1) * 16], 0, 0);
        }

        // ---- A fragment: two contiguous 8-half runs -> 2x ds_load_b128 ----
        // lanes 0-15: K 0..7 | 8..15 ; lanes 16-31: K 16..23 | 24..31
        const _Float16* arow = &lx[(m0 + n16) * 32];
        v8h a_lo = *(const v8h*)&arow[hi * 16];         // 16B aligned
        v8h a_hi = *(const v8h*)&arow[hi * 16 + 8];     // 16B aligned
        v16h af  = __builtin_shufflevector(a_lo, a_hi,
                     0, 1, 2, 3, 4, 5, 6, 7, 8, 9, 10, 11, 12, 13, 14, 15);

        // ---- 8 B fragments: one contiguous 16-half run each -> b128 loads ----
        // lane: col n = t*16 + lane%16, K = hi*16 .. hi*16+15 (sequential)
        for (int t = 0; t < 8; ++t) {
            int n = t * 16 + n16;
            v16h bf = *(const v16h*)&lwT[n * 32 + hi * 16];  // 32B aligned
            acc[t] = __builtin_amdgcn_wmma_f32_16x16x32_f16(
                /*neg_a=*/false, af, /*neg_b=*/false, bf,
                /*c_mod=*/(short)0, acc[t],
                /*reuse_a=*/false, /*reuse_b=*/false);
        }
        __syncthreads();
    }

    // ---- writeout: C/D layout VGPR v -> M = v + 8*hi, N = lane%16 ----
    for (int t = 0; t < 8; ++t) {
        int n = t * 16 + n16;
        for (int v = 0; v < 8; ++v) {
            int m = rowBase + m0 + v + hi * 8;
            if (m < N) h[(size_t)m * D_OUT + n] = acc[t][v];
        }
    }
}

// ---------------------------------------------------------------------------
// Kernel 2: f_src[i] = h[i,:].a[0:128], f_dst[i] = h[i,:].a[128:256]
// One wave per node; float4 loads; wave32 shuffle reduction.
// ---------------------------------------------------------------------------
__global__ __launch_bounds__(256) void k_proj(const float* __restrict__ h,
                                              const float* __restrict__ avec,
                                              float* __restrict__ f_src,
                                              float* __restrict__ f_dst, int N) {
    int wave = threadIdx.x >> 5, lane = threadIdx.x & 31;
    int node = blockIdx.x * 8 + wave;
    if (node >= N) return;
    const float4* hr = (const float4*)(h + (size_t)node * D_OUT);
    const float4* as = (const float4*)avec;            // a[0,   0:128)
    const float4* ad = (const float4*)(avec + D_OUT);  // a[0, 128:256)
    float4 hv = hr[lane];
    float4 s4 = as[lane];
    float4 d4 = ad[lane];
    float ps = hv.x * s4.x + hv.y * s4.y + hv.z * s4.z + hv.w * s4.w;
    float pd = hv.x * d4.x + hv.y * d4.y + hv.z * d4.z + hv.w * d4.w;
    for (int off = 16; off > 0; off >>= 1) {
        ps += __shfl_down(ps, off, 32);
        pd += __shfl_down(pd, off, 32);
    }
    if (lane == 0) { f_src[node] = ps; f_dst[node] = pd; }
}

// ---------------------------------------------------------------------------
// Kernel 3: per-edge attention weight + scatter-add (global f32 atomics).
// One wave per edge; each lane handles 4 consecutive output columns.
// ---------------------------------------------------------------------------
__global__ __launch_bounds__(256) void k_edge(const long long* __restrict__ edge,
                                              const float* __restrict__ h,
                                              const float* __restrict__ f_src,
                                              const float* __restrict__ f_dst,
                                              float* __restrict__ out,
                                              float* __restrict__ rowsum, int E) {
    int wave = threadIdx.x >> 5, lane = threadIdx.x & 31;
    long long e = (long long)blockIdx.x * 8 + wave;
    if (e >= E) return;
    int s = (int)edge[e];                 // src row
    int d = (int)edge[(size_t)E + e];     // dst row
    float tmp = f_src[s] + f_dst[d];
    float l   = tmp > 0.0f ? tmp : ALPHA * tmp;
    float w   = __expf(-l);
    const float4* hd = (const float4*)(h + (size_t)d * D_OUT);
    float4 hv = hd[lane];
    float* o = out + (size_t)s * D_OUT + lane * 4;
    atomicAdd(o + 0, w * hv.x);
    atomicAdd(o + 1, w * hv.y);
    atomicAdd(o + 2, w * hv.z);
    atomicAdd(o + 3, w * hv.w);
    if (lane == 0) atomicAdd(rowsum + s, w);
}

// ---------------------------------------------------------------------------
// Kernel 4: out = ELU(out / rowsum[row])
// ---------------------------------------------------------------------------
__global__ void k_final(float* __restrict__ out,
                        const float* __restrict__ rowsum, int total) {
    int stride = gridDim.x * blockDim.x;
    for (int t = blockIdx.x * blockDim.x + threadIdx.x; t < total; t += stride) {
        float v = out[t] / rowsum[t >> 7];
        out[t] = v > 0.0f ? v : expm1f(v);
    }
}

// ---------------------------------------------------------------------------
extern "C" void kernel_launch(void* const* d_in, const int* in_sizes, int n_in,
                              void* d_out, int out_size, void* d_ws, size_t ws_size,
                              hipStream_t stream) {
    const float*     x    = (const float*)d_in[0];      // [N, 256] f32
    const long long* edge = (const long long*)d_in[1];  // [2, E]   i64
    const float*     W    = (const float*)d_in[2];      // [256,128] f32
    const float*     a    = (const float*)d_in[3];      // [1, 256] f32

    const int N = in_sizes[0] / D_IN;
    const int E = in_sizes[1] / 2;

    float* out    = (float*)d_out;                      // [N,128] accum then ELU
    float* h      = (float*)d_ws;                       // N*128 f32 (51.2 MB)
    float* f_src  = h + (size_t)N * D_OUT;              // N
    float* f_dst  = f_src + N;                          // N
    float* rowsum = f_dst + N;                          // N

    k_zero <<<1024,            256, 0, stream>>>(out, rowsum, N * D_OUT, N);
    k_gemm <<<(N + 127) / 128, 256, 0, stream>>>(x, W, h, N);
    k_proj <<<(N + 7) / 8,     256, 0, stream>>>(h, a, f_src, f_dst, N);
    k_edge <<<(E + 7) / 8,     256, 0, stream>>>(edge, h, f_src, f_dst, out, rowsum, E);
    k_final<<<1024,            256, 0, stream>>>(out, rowsum, N * D_OUT);
}
